// TinyTransformerBlock_30717606101340
// MI455X (gfx1250) — matrix operand order repaired and checked by reading, not compiled, not run on hardware
//
#include <hip/hip_runtime.h>
#include <hip/hip_bf16.h>
#include <math.h>

typedef __attribute__((ext_vector_type(16))) _Float16 v16h;
typedef __attribute__((ext_vector_type(8)))  _Float16 v8h;
typedef __attribute__((ext_vector_type(8)))  float    v8f;

constexpr int CB  = 2;
constexpr int CT  = 2048;
constexpr int CM  = 2048;
constexpr int CD  = 384;
constexpr int CH  = 4;
constexpr int CDH = 96;
constexpr int CNM = 8;
constexpr int CDF = 1536;
constexpr int CR  = 16;
constexpr int CRP = 32;                 // LoRA rank padded to one WMMA K-step
constexpr int NROWS = CB * CT;          // 4096 (== CB*CM)
constexpr float SM_SCALE = 0.1020620726159657f; // 1/sqrt(96)

#define DEVFN static __device__ __forceinline__

// ---- WMMA fragment loaders (contiguous k = 16*(lane/16) + s in both A and B, 16-bit, wave32) ----
// A (16xK=32, MxK): lanes 0-15 -> M=lane, lanes 16-31 -> M=lane-16.
// Per lane: elements 0..15 = K=kb..kb+15, kb = (lane>=16)?16:0
DEVFN v16h load_a_frag(const _Float16* p0, int lda, int lid) {
  const int m  = lid & 15;
  const int kb = (lid >> 4) * 16;
  const _Float16* p = p0 + (size_t)m * lda + kb;
  v8h lo = *(const v8h*)(p);
  v8h hi = *(const v8h*)(p + 8);
  v16h r;
#pragma unroll
  for (int i = 0; i < 8; ++i) { r[i] = lo[i]; r[8 + i] = hi[i]; }
  return r;
}

// B (K=32 x N=16): lanes 0-15 -> N=lane, K=0..15; lanes 16-31 -> N=lane-16, K=16..31.
// With B(k,n) = W[n0+n][k0+k] (W row-major over K) each lane reads 16 contiguous halves.
DEVFN v16h load_b_frag(const _Float16* p0, int ldw, int lid) {
  const int n  = lid & 15;
  const int kb = (lid >> 4) * 16;
  return *(const v16h*)(p0 + (size_t)n * ldw + kb);
}

// Same B layout, but from an LDS tile shaped [64 cols][32 k] halves.
DEVFN v16h load_b_frag_lds(const _Float16* tile, int lid, int f) {
  const int n  = lid & 15;
  const int kb = (lid >> 4) * 16;
  return *(const v16h*)(tile + (size_t)(f * 16 + n) * 32 + kb);
}

DEVFN v8f wmma_f16(v16h a, v16h b, v8f c) {
  return __builtin_amdgcn_wmma_f32_16x16x32_f16(false, a, false, b, (short)0, c, false, false);
}

// Async global->LDS cooperative fill of a 64x32-half W tile (4KB, 16B/thread,
// 256 threads). One GLOBAL_LOAD_ASYNC_TO_LDS_B128 per wave, tracked by ASYNCcnt.
DEVFN void async_fill_b(const _Float16* W, int ldw, int col0, int k0,
                        unsigned lds_base, int t) {
  const int row = t >> 2;  // 0..63 (tile col)
  const int seg = t & 3;   // 16B segment within the 64B row
  const unsigned voff =
      (unsigned)(((size_t)(col0 + row) * (size_t)ldw + (size_t)(k0 + seg * 8)) * sizeof(_Float16));
  const unsigned lds = lds_base + (unsigned)(row * 64 + seg * 16);
  asm volatile("global_load_async_to_lds_b128 %0, %1, %2 offset:0"
               :: "v"(lds), "v"(voff), "s"(W) : "memory");
}

// --------------------------- generic WMMA GEMM -------------------------------
// C[row,col] = sum_k A[row,k] * W[col,k] (+bias[col]) (+resid) ; epilogue by MODE
// MODE 0: f16 out ; MODE 1: f32 out + residual ; MODE 2: gelu -> f16 out
// CFRAGS==4: W tile double-buffered in LDS via async loads (shared by 8 waves).
template <int MODE, int CFRAGS>
__global__ __launch_bounds__(256) void ttb_gemm(
    const _Float16* __restrict__ A, int lda,
    const _Float16* __restrict__ W, int ldw,
    const float* __restrict__ bias,
    const float* __restrict__ resid, int ldr,
    float* __restrict__ outf, _Float16* __restrict__ outh, int ldo, int Ktot) {
  const int lid  = threadIdx.x & 31;
  const int widx = threadIdx.x >> 5;
  const int row0 = (blockIdx.x * 8 + widx) * 16;
  const int col0 = blockIdx.y * (16 * CFRAGS);

  v8f acc[CFRAGS];
#pragma unroll
  for (int f = 0; f < CFRAGS; ++f) acc[f] = (v8f)0.0f;

  if constexpr (CFRAGS == 4) {
    __shared__ __align__(64) _Float16 Btile[2][64][32];
    const unsigned lds0 = (unsigned)(uintptr_t)&Btile[0][0][0];
    const unsigned lds1 = (unsigned)(uintptr_t)&Btile[1][0][0];
    const int t  = threadIdx.x;
    const int KC = Ktot / 32;

    async_fill_b(W, ldw, col0, 0, lds0, t);
    for (int kc = 0; kc < KC; ++kc) {
      const int k0 = kc * 32;
      if (kc + 1 < KC) {
        async_fill_b(W, ldw, col0, k0 + 32, ((kc + 1) & 1) ? lds1 : lds0, t);
        asm volatile("s_wait_asynccnt 0x1" ::: "memory");
      } else {
        asm volatile("s_wait_asynccnt 0x0" ::: "memory");
      }
      __syncthreads();  // tile kc resident for the whole block
      v16h a = load_a_frag(A + (size_t)row0 * lda + k0, lda, lid);
      const _Float16* bt = &Btile[kc & 1][0][0];
#pragma unroll
      for (int f = 0; f < CFRAGS; ++f) {
        v16h bw = load_b_frag_lds(bt, lid, f);
        acc[f] = wmma_f16(a, bw, acc[f]);
      }
      __syncthreads();  // all reads done before this buffer is refilled
    }
  } else {
    for (int k0 = 0; k0 < Ktot; k0 += 32) {
      v16h a = load_a_frag(A + (size_t)row0 * lda + k0, lda, lid);
#pragma unroll
      for (int f = 0; f < CFRAGS; ++f) {
        v16h bw = load_b_frag(W + (size_t)(col0 + f * 16) * ldw + k0, ldw, lid);
        acc[f] = wmma_f16(a, bw, acc[f]);
      }
    }
  }

  const int n  = lid & 15;
  const int rb = (lid >> 4) * 8;
#pragma unroll
  for (int f = 0; f < CFRAGS; ++f) {
#pragma unroll
    for (int r = 0; r < 8; ++r) {
      const int row = row0 + rb + r;
      const int col = col0 + f * 16 + n;
      float v = acc[f][r];
      if (bias) v += bias[col];
      if (MODE == 0) {
        outh[(size_t)row * ldo + col] = (_Float16)v;
      } else if (MODE == 1) {
        outf[(size_t)row * ldo + col] = v + resid[(size_t)row * ldr + col];
      } else {
        float g = 0.5f * v * (1.0f + erff(v * 0.70710678118654752f)); // exact gelu
        outh[(size_t)row * ldo + col] = (_Float16)g;
      }
    }
  }
}

// ----------------------- flash attention (WMMA, wave32) ----------------------
// One wave owns 16 query rows of one (b,h). KV consumed in blocks of 32.
// Z = gate * softmax(Q K^T * scale) V ; Vt is [B,H,DH,M] so P*V B-frags are contiguous.
__global__ __launch_bounds__(128) void ttb_attn(
    const _Float16* __restrict__ Q, const _Float16* __restrict__ K,
    const _Float16* __restrict__ Vt, const float* __restrict__ gates,
    _Float16* __restrict__ Z) {
  __shared__ __align__(32) _Float16 Pbuf[4][16][32];
  const int lid  = threadIdx.x & 31;
  const int widx = threadIdx.x >> 5;
  const int gw   = blockIdx.x * 4 + widx;
  const int qt   = gw % (CT / 16);
  const int bh   = gw / (CT / 16);
  const int b    = bh / CH;
  const int h    = bh % CH;

  const _Float16* Qb  = Q + ((size_t)(b * CT + qt * 16)) * CD + h * CDH;
  const _Float16* Kb  = K + (size_t)b * CM * CD + h * CDH;
  const _Float16* Vtb = Vt + ((size_t)(b * CH + h)) * CDH * CM;

  // Q fragments (3 K-chunks of 32 over DH=96), pre-scaled by 1/sqrt(DH)
  v16h aq[3];
#pragma unroll
  for (int c = 0; c < 3; ++c) {
    v16h t = load_a_frag(Qb + c * 32, CD, lid);
#pragma unroll
    for (int i = 0; i < 16; ++i) t[i] = (_Float16)((float)t[i] * SM_SCALE);
    aq[c] = t;
  }

  float mprev[8], lsum[8];
  v8f o[6];
#pragma unroll
  for (int r = 0; r < 8; ++r) { mprev[r] = -1e30f; lsum[r] = 0.0f; }
#pragma unroll
  for (int f = 0; f < 6; ++f) o[f] = (v8f)0.0f;

  const int nn = lid & 15;
  const int rb = (lid >> 4) * 8;

  for (int m0 = 0; m0 < CM; m0 += 32) {
    v8f s0 = (v8f)0.0f, s1 = (v8f)0.0f;
#pragma unroll
    for (int c = 0; c < 3; ++c) {
      v16h bk0 = load_b_frag(Kb + (size_t)m0 * CD + c * 32, CD, lid);
      v16h bk1 = load_b_frag(Kb + (size_t)(m0 + 16) * CD + c * 32, CD, lid);
      s0 = wmma_f16(aq[c], bk0, s0);
      s1 = wmma_f16(aq[c], bk1, s1);
    }
    // block row-max (reduce across the 16 lanes of each half-wave)
    float mloc[8];
#pragma unroll
    for (int r = 0; r < 8; ++r) mloc[r] = fmaxf(s0[r], s1[r]);
#pragma unroll
    for (int off = 1; off <= 8; off <<= 1)
#pragma unroll
      for (int r = 0; r < 8; ++r)
        mloc[r] = fmaxf(mloc[r], __shfl_xor(mloc[r], off, 32));

    float p0a[8], p1a[8], psum[8];
#pragma unroll
    for (int r = 0; r < 8; ++r) {
      float mnew = fmaxf(mprev[r], mloc[r]);
      float corr = __expf(mprev[r] - mnew);
      p0a[r] = __expf(s0[r] - mnew);
      p1a[r] = __expf(s1[r] - mnew);
      psum[r] = p0a[r] + p1a[r];
      lsum[r] *= corr;
      mprev[r] = mnew;
#pragma unroll
      for (int f = 0; f < 6; ++f) o[f][r] *= corr;
    }
#pragma unroll
    for (int off = 1; off <= 8; off <<= 1)
#pragma unroll
      for (int r = 0; r < 8; ++r) psum[r] += __shfl_xor(psum[r], off, 32);
#pragma unroll
    for (int r = 0; r < 8; ++r) lsum[r] += psum[r];

    // C-layout -> A-layout through per-wave LDS tile
#pragma unroll
    for (int r = 0; r < 8; ++r) {
      Pbuf[widx][rb + r][nn]      = (_Float16)p0a[r];
      Pbuf[widx][rb + r][16 + nn] = (_Float16)p1a[r];
    }
    asm volatile("s_wait_dscnt 0x0" ::: "memory");
    v16h ap = load_a_frag(&Pbuf[widx][0][0], 32, lid);
#pragma unroll
    for (int f = 0; f < 6; ++f) {
      v16h bv = load_b_frag(Vtb + (size_t)(f * 16) * CM + m0, CM, lid);
      o[f] = wmma_f16(ap, bv, o[f]);
    }
    asm volatile("s_wait_dscnt 0x0" ::: "memory");
  }

  float g = 1.0f;
  if (gates) g = gates[b * CH + h];
#pragma unroll
  for (int r = 0; r < 8; ++r) {
    const float inv = g / lsum[r];
    const size_t row = (size_t)(b * CT + qt * 16 + rb + r);
#pragma unroll
    for (int f = 0; f < 6; ++f)
      Z[row * CD + h * CDH + f * 16 + nn] = (_Float16)(o[f][r] * inv);
  }
}

// ------------------------------- LayerNorm -----------------------------------
__global__ __launch_bounds__(128) void ttb_layernorm(
    const float* __restrict__ x, const float* __restrict__ gam,
    const float* __restrict__ bet, _Float16* __restrict__ out) {
  __shared__ float rs1[128], rs2[128];
  const int row = blockIdx.x;
  const int tid = threadIdx.x;
  const float* xr = x + (size_t)row * CD;
  float v[3], s = 0.f, ss = 0.f;
#pragma unroll
  for (int k = 0; k < 3; ++k) { v[k] = xr[tid + k * 128]; s += v[k]; ss += v[k] * v[k]; }
  rs1[tid] = s; rs2[tid] = ss;
  __syncthreads();
  for (int st = 64; st > 0; st >>= 1) {
    if (tid < st) { rs1[tid] += rs1[tid + st]; rs2[tid] += rs2[tid + st]; }
    __syncthreads();
  }
  const float mu   = rs1[0] * (1.0f / CD);
  const float var  = rs2[0] * (1.0f / CD) - mu * mu;
  const float rstd = rsqrtf(var + 1e-5f);
#pragma unroll
  for (int k = 0; k < 3; ++k) {
    const int i = tid + k * 128;
    out[(size_t)row * CD + i] = (_Float16)((v[k] - mu) * rstd * gam[i] + bet[i]);
  }
}

// ------------------------------ small helpers --------------------------------
__global__ void ttb_cvt(const float* __restrict__ s, _Float16* __restrict__ d, int n) {
  for (int i = blockIdx.x * blockDim.x + threadIdx.x; i < n; i += gridDim.x * blockDim.x)
    d[i] = (_Float16)s[i];
}
__global__ void ttb_cvt_slice(const float* __restrict__ base, const int* __restrict__ idx,
                              int slice, _Float16* __restrict__ d, int n) {
  const float* s = base + (size_t)idx[0] * slice;
  for (int i = blockIdx.x * blockDim.x + threadIdx.x; i < n; i += gridDim.x * blockDim.x)
    d[i] = (_Float16)s[i];
}
__global__ void ttb_cvt_padB(const float* __restrict__ base, const int* __restrict__ idx,
                             _Float16* __restrict__ d) {  // adB[idx]: [CD,16] -> [CD,32] zero-pad
  const float* s = base + (size_t)idx[0] * (CD * CR);
  int i = blockIdx.x * blockDim.x + threadIdx.x;
  if (i < CD * CRP) {
    int row = i / CRP, k = i % CRP;
    d[i] = (k < CR) ? (_Float16)s[row * CR + k] : (_Float16)0.0f;
  }
}
__global__ void ttb_zero16(_Float16* p, int n) {
  for (int i = blockIdx.x * blockDim.x + threadIdx.x; i < n; i += gridDim.x * blockDim.x)
    p[i] = (_Float16)0.0f;
}
__global__ void ttb_vtrans(const _Float16* __restrict__ v, _Float16* __restrict__ vt) {
  int i = blockIdx.x * blockDim.x + threadIdx.x;
  if (i >= CB * CM * CD) return;
  int dcol = i % CD, rm = i / CD;
  int b = rm / CM, m = rm % CM;
  int h = dcol / CDH, dh = dcol % CDH;
  vt[(((size_t)(b * CH + h)) * CDH + dh) * CM + m] = v[i];
}
__global__ void ttb_gates(const float* __restrict__ motif, const float* __restrict__ gW,
                          const float* __restrict__ gb, float* __restrict__ gout) {
  int t = threadIdx.x;
  if (t < CB * CH) {
    int b = t / CH, h = t % CH;
    float s = gb[h];
    for (int j = 0; j < CNM; ++j) s += motif[b * CNM + j] * gW[h * CNM + j];
    gout[t] = 1.0f / (1.0f + __expf(-s));
  }
}

// --------------------------------- launcher ----------------------------------
extern "C" void kernel_launch(void* const* d_in, const int* in_sizes, int n_in,
                              void* d_out, int out_size, void* d_ws, size_t ws_size,
                              hipStream_t stream) {
  (void)in_sizes; (void)n_in; (void)out_size; (void)ws_size;
  const float* x     = (const float*)d_in[0];
  const float* mem   = (const float*)d_in[1];
  const float* motif = (const float*)d_in[2];
  const int*   aidx  = (const int*)d_in[3];
  const float* cWq = (const float*)d_in[4],  *cbq = (const float*)d_in[5];
  const float* cWk = (const float*)d_in[6],  *cbk = (const float*)d_in[7];
  const float* cWv = (const float*)d_in[8],  *cbv = (const float*)d_in[9];
  const float* cWo = (const float*)d_in[10], *cbo = (const float*)d_in[11];
  const float* sWq = (const float*)d_in[12], *sbq = (const float*)d_in[13];
  const float* sWk = (const float*)d_in[14], *sbk = (const float*)d_in[15];
  const float* sWv = (const float*)d_in[16], *sbv = (const float*)d_in[17];
  const float* sWo = (const float*)d_in[18], *sbo = (const float*)d_in[19];
  const float* gW  = (const float*)d_in[20], *gb  = (const float*)d_in[21];
  const float* f1W = (const float*)d_in[22], *f1b = (const float*)d_in[23];
  const float* f2W = (const float*)d_in[24], *f2b = (const float*)d_in[25];
  const float* adA = (const float*)d_in[26], *adB = (const float*)d_in[27];
  const float* ln1g = (const float*)d_in[28], *ln1b = (const float*)d_in[29];
  const float* ln2g = (const float*)d_in[30], *ln2b = (const float*)d_in[31];
  const float* ln3g = (const float*)d_in[32], *ln3b = (const float*)d_in[33];
  float* out = (float*)d_out;

  char* base = (char*)d_ws;
  size_t off = 0;
  auto carve = [&](size_t bytes) -> void* {
    void* p = base + off;
    off = (off + bytes + 255) & ~(size_t)255;
    return p;
  };
  auto c16 = [&](size_t n) { return (_Float16*)carve(n * sizeof(_Float16)); };
  auto c32 = [&](size_t n) { return (float*)carve(n * sizeof(float)); };

  _Float16 *wq16 = c16(CD * CD), *wk16 = c16(CD * CD), *wv16 = c16(CD * CD), *wo16 = c16(CD * CD);
  _Float16 *vq16 = c16(CD * CD), *vk16 = c16(CD * CD), *vv16 = c16(CD * CD), *vo16 = c16(CD * CD);
  _Float16 *f1w16 = c16((size_t)CDF * CD), *f2w16 = c16((size_t)CD * CDF);
  _Float16 *adA16 = c16(CR * CD), *adB16 = c16(CD * CRP);
  float    *gatesd = c32(CB * CH);
  _Float16 *mem16 = c16((size_t)NROWS * CD);
  _Float16 *h1 = c16((size_t)NROWS * CD), *h2 = c16((size_t)NROWS * CD), *h3 = c16((size_t)NROWS * CD);
  _Float16 *qb = c16((size_t)NROWS * CD), *kbuf = c16((size_t)NROWS * CD);
  _Float16 *vb = c16((size_t)NROWS * CD), *vtb = c16((size_t)NROWS * CD);
  _Float16 *zb = c16((size_t)NROWS * CD);
  float    *x1 = c32((size_t)NROWS * CD), *x2 = c32((size_t)NROWS * CD), *t1 = c32((size_t)NROWS * CD);
  _Float16 *gact = c16((size_t)NROWS * CDF);
  _Float16 *lr = c16((size_t)NROWS * CRP);

  // weight conversions (f32 -> f16)
  const float* wsrc[10] = {cWq, cWk, cWv, cWo, sWq, sWk, sWv, sWo, f1W, f2W};
  _Float16*    wdst[10] = {wq16, wk16, wv16, wo16, vq16, vk16, vv16, vo16, f1w16, f2w16};
  const int    wcnt[10] = {CD*CD, CD*CD, CD*CD, CD*CD, CD*CD, CD*CD, CD*CD, CD*CD, CDF*CD, CD*CDF};
  for (int i = 0; i < 10; ++i)
    ttb_cvt<<<dim3(256), dim3(256), 0, stream>>>(wsrc[i], wdst[i], wcnt[i]);
  ttb_cvt<<<dim3(1024), dim3(256), 0, stream>>>(mem, mem16, NROWS * CD);
  ttb_cvt_slice<<<dim3(24), dim3(256), 0, stream>>>(adA, aidx, CR * CD, adA16, CR * CD);
  ttb_cvt_padB<<<dim3((CD * CRP + 255) / 256), dim3(256), 0, stream>>>(adB, aidx, adB16);
  ttb_gates<<<dim3(1), dim3(32), 0, stream>>>(motif, gW, gb, gatesd);

  const dim3 gD(NROWS / 128, CD / 64);     // GEMMs with N_out = 384
  const dim3 gF1(NROWS / 128, CDF / 64);   // N_out = 1536
  const dim3 gLr(NROWS / 128, 1);          // N_out = 16
  const dim3 blk(256);

  // ---- cross attention ----
  ttb_layernorm<<<dim3(NROWS), dim3(128), 0, stream>>>(x, ln1g, ln1b, h1);
  ttb_gemm<0, 4><<<gD, blk, 0, stream>>>(h1, CD, wq16, CD, cbq, nullptr, 0, nullptr, qb, CD, CD);
  ttb_gemm<0, 4><<<gD, blk, 0, stream>>>(mem16, CD, wk16, CD, cbk, nullptr, 0, nullptr, kbuf, CD, CD);
  ttb_gemm<0, 4><<<gD, blk, 0, stream>>>(mem16, CD, wv16, CD, cbv, nullptr, 0, nullptr, vb, CD, CD);
  ttb_vtrans<<<dim3((CB * CM * CD + 255) / 256), dim3(256), 0, stream>>>(vb, vtb);
  ttb_attn<<<dim3(CB * CH * (CT / 16) / 4), dim3(128), 0, stream>>>(qb, kbuf, vtb, nullptr, zb);
  ttb_gemm<1, 4><<<gD, blk, 0, stream>>>(zb, CD, wo16, CD, cbo, x, CD, x1, nullptr, CD, CD);

  // ---- gated self attention ----
  ttb_layernorm<<<dim3(NROWS), dim3(128), 0, stream>>>(x1, ln2g, ln2b, h2);
  ttb_gemm<0, 4><<<gD, blk, 0, stream>>>(h2, CD, vq16, CD, sbq, nullptr, 0, nullptr, qb, CD, CD);
  ttb_gemm<0, 4><<<gD, blk, 0, stream>>>(h2, CD, vk16, CD, sbk, nullptr, 0, nullptr, kbuf, CD, CD);
  ttb_gemm<0, 4><<<gD, blk, 0, stream>>>(h2, CD, vv16, CD, sbv, nullptr, 0, nullptr, vb, CD, CD);
  ttb_vtrans<<<dim3((CB * CM * CD + 255) / 256), dim3(256), 0, stream>>>(vb, vtb);
  ttb_attn<<<dim3(CB * CH * (CT / 16) / 4), dim3(128), 0, stream>>>(qb, kbuf, vtb, gatesd, zb);
  ttb_gemm<1, 4><<<gD, blk, 0, stream>>>(zb, CD, vo16, CD, sbo, x1, CD, x2, nullptr, CD, CD);

  // ---- FFN + LoRA ----
  ttb_layernorm<<<dim3(NROWS), dim3(128), 0, stream>>>(x2, ln3g, ln3b, h3);
  ttb_gemm<2, 4><<<gF1, blk, 0, stream>>>(h3, CD, f1w16, CD, f1b, nullptr, 0, nullptr, gact, CDF, CD);
  ttb_gemm<1, 4><<<gD, blk, 0, stream>>>(gact, CDF, f2w16, CDF, f2b, x2, CD, t1, nullptr, CD, CDF);
  ttb_zero16<<<dim3(512), dim3(256), 0, stream>>>(lr, NROWS * CRP);
  ttb_gemm<0, 1><<<gLr, blk, 0, stream>>>(h3, CD, adA16, CD, nullptr, nullptr, 0, nullptr, lr, CRP, CD);
  ttb_gemm<1, 4><<<gD, blk, 0, stream>>>(lr, CRP, adB16, CRP, nullptr, t1, CD, out, nullptr, CD, CRP);
}
